// CausalMultiHeadSelfAttention_1992864825562
// MI455X (gfx1250) — matrix operand order repaired and checked by reading, not compiled, not run on hardware
//
#include <hip/hip_runtime.h>
#include <hip/hip_bf16.h>

typedef __bf16 bf16;
typedef __attribute__((ext_vector_type(16))) __bf16 v16bf;
typedef __attribute__((ext_vector_type(8)))  __bf16 v8bf;
typedef __attribute__((ext_vector_type(4)))  __bf16 v4bf;
typedef __attribute__((ext_vector_type(8)))  float   v8f;
typedef __attribute__((ext_vector_type(4)))  float   v4f;
typedef int v4i __attribute__((vector_size(16)));   // matches async-LDS builtin param

static constexpr int BB = 2;
static constexpr int T  = 2048;
static constexpr int D  = 1024;
static constexpr int H  = 16;
static constexpr int DK = 64;
static constexpr int M  = BB * T;          // 4096 rows in all GEMMs
static constexpr float SCALE = 0.125f;     // 1/sqrt(dk)

#define WMMA_BF16(a, b, c) \
  __builtin_amdgcn_wmma_f32_16x16x32_bf16(false, (a), false, (b), (short)0, (c), false, false)

// gfx1250 async global->LDS path (ASYNCcnt). Guarded so the file compiles on
// toolchains lacking the builtins (falls back to direct global fragment loads).
#if defined(__has_builtin)
# if __has_builtin(__builtin_amdgcn_global_load_async_to_lds_b128) && \
     __has_builtin(__builtin_amdgcn_s_wait_asynccnt)
#  define USE_ASYNC_LDS 1
# endif
#endif
#ifndef USE_ASYNC_LDS
# define USE_ASYNC_LDS 0
#endif

#if USE_ASYNC_LDS
__device__ __forceinline__ void async_copy16(const bf16* g, bf16* l) {
  __builtin_amdgcn_global_load_async_to_lds_b128(
      (__attribute__((address_space(1))) v4i*)(bf16*)g,
      (__attribute__((address_space(3))) v4i*)l,
      0, 0);
}
#endif

// ---------------------------------------------------------------------------
// Fragment loaders for wave32 WMMA: contiguous k order shared by A and B (not AMD's table order)
// A (16x32 bf16): lane m = lane&15, half = lane>>4.
//   VGPR0-7 hold K = half*16 + 0..15 (contiguous).
// ---------------------------------------------------------------------------
__device__ __forceinline__ v16bf load_a_frag(const bf16* __restrict__ base, int ld) {
  const int lane = threadIdx.x & 31;
  const int m = lane & 15;
  const int hf = lane >> 4;
  const bf16* p = base + (size_t)m * ld + hf * 16;
  v8bf lo = *reinterpret_cast<const v8bf*>(p);
  v8bf hi = *reinterpret_cast<const v8bf*>(p + 8);
  v16bf r;
#pragma unroll
  for (int i = 0; i < 8; ++i) { r[i] = lo[i]; r[8 + i] = hi[i]; }
  return r;
}

// B (32x16 bf16): lane n = lane&15, half = lane>>4 selects K window (0-15/16-31),
//   VGPR v holds K = half*16 + 2v,2v+1 -> one contiguous 32B run per lane.
__device__ __forceinline__ v16bf load_b_frag(const bf16* __restrict__ base, int ld) {
  const int lane = threadIdx.x & 31;
  const int n = lane & 15;
  const int hf = lane >> 4;
  const bf16* p = base + (size_t)n * ld + hf * 16;
  v8bf lo = *reinterpret_cast<const v8bf*>(p);
  v8bf hi = *reinterpret_cast<const v8bf*>(p + 8);
  v16bf r;
#pragma unroll
  for (int i = 0; i < 8; ++i) { r[i] = lo[i]; r[8 + i] = hi[i]; }
  return r;
}

// ---------------------------------------------------------------------------
// fp32 -> bf16 conversion, 4 elements per thread
// ---------------------------------------------------------------------------
__global__ __launch_bounds__(256) void cvt_kernel(const float* __restrict__ src,
                                                  bf16* __restrict__ dst, int n4) {
  const int i = blockIdx.x * blockDim.x + threadIdx.x;
  if (i < n4) {
    v4f f = reinterpret_cast<const v4f*>(src)[i];
    v4bf o;
#pragma unroll
    for (int j = 0; j < 4; ++j) o[j] = (bf16)f[j];
    reinterpret_cast<v4bf*>(dst)[i] = o;
  }
}

// ---------------------------------------------------------------------------
// WMMA GEMM:  C[m,n] = sum_k A[m,k] * Wt[n,k]   (A: [M,D] bf16, Wt: [D,D] bf16)
// One wave computes a 64x64 C tile (16 WMMAs per 32-wide K step; 1.0 b128
// loads per WMMA). mode: 0 -> bf16 [B,H,T,dk]; 1 -> bf16 [B,H,dk,T] (v);
// 2 -> fp32 row-major [M,D] (final output).
// ---------------------------------------------------------------------------
__global__ __launch_bounds__(256) void gemm_kernel(const bf16* __restrict__ A,
                                                   const bf16* __restrict__ Wt,
                                                   bf16* __restrict__ Cb,
                                                   float* __restrict__ Cf,
                                                   int mode) {
  const int wave  = blockIdx.x * (blockDim.x >> 5) + (threadIdx.x >> 5);
  const int NT    = D / 64;           // 16 n-groups
  const int mt    = wave / NT;        // 0..63
  const int ng    = wave % NT;
  const int mbase = mt * 64;
  const int nbase = ng * 64;

  v8f zero = {};
  v8f acc[4][4];
#pragma unroll
  for (int mi = 0; mi < 4; ++mi)
#pragma unroll
    for (int nt = 0; nt < 4; ++nt) acc[mi][nt] = zero;

  for (int k0 = 0; k0 < D; k0 += 32) {
    v16bf a[4];
#pragma unroll
    for (int mi = 0; mi < 4; ++mi)
      a[mi] = load_a_frag(A + (size_t)(mbase + mi * 16) * D + k0, D);
#pragma unroll
    for (int nt = 0; nt < 4; ++nt) {
      v16bf b = load_b_frag(Wt + (size_t)(nbase + nt * 16) * D + k0, D);
#pragma unroll
      for (int mi = 0; mi < 4; ++mi)
        acc[mi][nt] = WMMA_BF16(a[mi], b, acc[mi][nt]);
    }
  }

  // C layout (16x16 f32): lane n = lane&15; lanes 0-15 rows 0-7, lanes 16-31 rows 8-15
  const int lane = threadIdx.x & 31;
  const int n16  = lane & 15;
  const int hf   = lane >> 4;
#pragma unroll
  for (int mi = 0; mi < 4; ++mi)
#pragma unroll
    for (int nt = 0; nt < 4; ++nt)
#pragma unroll
      for (int r = 0; r < 8; ++r) {
        const int m  = mbase + mi * 16 + hf * 8 + r;
        const int nc = nbase + nt * 16 + n16;
        const float val = acc[mi][nt][r];
        if (mode == 2) {
          Cf[(size_t)m * D + nc] = val;
        } else {
          const int b = m >> 11;        // / T
          const int t = m & (T - 1);
          const int h = nc >> 6;        // / dk
          const int d = nc & (DK - 1);
          const bf16 bv = (bf16)val;
          if (mode == 0)
            Cb[((size_t)(b * H + h) * T + t) * DK + d] = bv;        // [B,H,T,dk]
          else
            Cb[((size_t)(b * H + h) * DK + d) * T + t] = bv;        // [B,H,dk,T]
        }
      }
}

// ---------------------------------------------------------------------------
// RoPE in place on q and k, layout [B*H, T, dk], bf16
// ---------------------------------------------------------------------------
__global__ __launch_bounds__(256) void rope_kernel(bf16* __restrict__ Q,
                                                   bf16* __restrict__ K) {
  const int idx = blockIdx.x * blockDim.x + threadIdx.x; // B*H*T*32 threads
  const int i   = idx & 31;                 // pair index within dk/2
  const int t   = (idx >> 5) & (T - 1);
  const int bh  = idx >> 16;                // / (32*T)
  const float inv = __expf(-((float)(2 * i) / (float)DK) * 6.907755278982137f); // ln(1000)
  const float ang = (float)t * inv;
  const float c = cosf(ang);
  const float s = sinf(ang);
  const size_t base = ((size_t)bh * T + t) * DK + 2 * i;
  {
    const float xe = (float)Q[base], xo = (float)Q[base + 1];
    Q[base]     = (bf16)(xe * c - xo * s);
    Q[base + 1] = (bf16)(xe * s + xo * c);
  }
  {
    const float xe = (float)K[base], xo = (float)K[base + 1];
    K[base]     = (bf16)(xe * c - xo * s);
    K[base + 1] = (bf16)(xe * s + xo * c);
  }
}

// ---------------------------------------------------------------------------
// Flash attention: one wave per (b,h, 16-query tile); 32-key blocks.
// Q,K: [B*H, T, dk] bf16 ; Vt: [B*H, dk, T] bf16 ; C: [B, T, D] bf16 concat.
// K/V tiles are double-buffered into LDS with async global->LDS copies
// (ASYNCcnt), prefetching block j+1 while block j computes.
// ---------------------------------------------------------------------------
__global__ __launch_bounds__(32) void attn_kernel(const bf16* __restrict__ Q,
                                                  const bf16* __restrict__ K,
                                                  const bf16* __restrict__ Vt,
                                                  bf16* __restrict__ C) {
  const int qt = blockIdx.x & (T / 16 - 1);   // 0..127
  const int bh = blockIdx.x >> 7;             // 0..31
  const int lane = threadIdx.x & 31;
  const int n16  = lane & 15;
  const int hf   = lane >> 4;

  const bf16* qbase = Q + ((size_t)bh * T + qt * 16) * DK;
  const bf16* kbase = K + (size_t)bh * T * DK;
  const bf16* vbase = Vt + (size_t)bh * DK * T;

  const v16bf qf0 = load_a_frag(qbase, DK);        // dk 0..31
  const v16bf qf1 = load_a_frag(qbase + 32, DK);   // dk 32..63

  __shared__ __align__(16) bf16 pLds[16 * 32];

  float mrow[8], lrow[8];
#pragma unroll
  for (int r = 0; r < 8; ++r) { mrow[r] = -1e30f; lrow[r] = 0.f; }
  v8f zero = {};
  v8f o[4] = {zero, zero, zero, zero};

  const int nb = (qt >> 1) + 1;               // 32-key blocks needed (causal)

#if USE_ASYNC_LDS
  __shared__ __align__(16) bf16 kT[2][32 * DK];   // [key 0..31][dk 0..63]
  __shared__ __align__(16) bf16 vT[2][DK * 32];   // [dk 0..63][key 0..31]
  auto prefetch = [&](int jb, int buf) {
    const int key0 = jb * 32;
    // K tile: 32 contiguous rows of 64 bf16 = 4096 contiguous bytes
    const bf16* ksrc = kbase + (size_t)key0 * DK;
#pragma unroll
    for (int i = 0; i < 8; ++i) {
      const int off = (i * 32 + lane) * 8;        // bf16 units, 16B chunks
      async_copy16(ksrc + off, &kT[buf][off]);
    }
    // V tile: 64 rows (dk), 64B each, row stride T*2 bytes in global
#pragma unroll
    for (int i = 0; i < 8; ++i) {
      const int row = i * 8 + (lane >> 2);
      const int ch  = (lane & 3) * 8;
      async_copy16(vbase + (size_t)row * T + key0 + ch, &vT[buf][row * 32 + ch]);
    }
  };
  prefetch(0, 0);
#endif

  for (int jb = 0; jb < nb; ++jb) {
    const int key0 = jb * 32;

    // ---- acquire K/V fragments for this key block ----
    v8f s0 = zero, s1 = zero;
#if USE_ASYNC_LDS
    const int cur = jb & 1;
    if (jb + 1 < nb) {
      prefetch(jb + 1, cur ^ 1);
      __builtin_amdgcn_s_wait_asynccnt(16);   // batch jb landed in LDS
    } else {
      __builtin_amdgcn_s_wait_asynccnt(0);
    }
    asm volatile("" ::: "memory");
    const bf16* kk = &kT[cur][0];
    const bf16* vv = &vT[cur][0];
    {
      v16bf bk = load_b_frag(kk, DK);
      s0 = WMMA_BF16(qf0, bk, s0);
      bk = load_b_frag(kk + 32, DK);
      s0 = WMMA_BF16(qf1, bk, s0);
      bk = load_b_frag(kk + 16 * DK, DK);
      s1 = WMMA_BF16(qf0, bk, s1);
      bk = load_b_frag(kk + 16 * DK + 32, DK);
      s1 = WMMA_BF16(qf1, bk, s1);
    }
#else
    {
      v16bf bk = load_b_frag(kbase + (size_t)key0 * DK, DK);
      s0 = WMMA_BF16(qf0, bk, s0);
      bk = load_b_frag(kbase + (size_t)key0 * DK + 32, DK);
      s0 = WMMA_BF16(qf1, bk, s0);
      bk = load_b_frag(kbase + (size_t)(key0 + 16) * DK, DK);
      s1 = WMMA_BF16(qf0, bk, s1);
      bk = load_b_frag(kbase + (size_t)(key0 + 16) * DK + 32, DK);
      s1 = WMMA_BF16(qf1, bk, s1);
    }
#endif

    // ---- online softmax (rows hf*8+r in this lane; columns across 16 lanes) ----
    float p0v[8], p1v[8], alpha[8];
#pragma unroll
    for (int r = 0; r < 8; ++r) {
      const int qrow = qt * 16 + hf * 8 + r;
      float a = s0[r] * SCALE;
      a = (key0 + n16 > qrow) ? -1e30f : a;
      float b = s1[r] * SCALE;
      b = (key0 + 16 + n16 > qrow) ? -1e30f : b;
      float mx = fmaxf(a, b);
      mx = fmaxf(mx, __shfl_xor(mx, 8, 16));
      mx = fmaxf(mx, __shfl_xor(mx, 4, 16));
      mx = fmaxf(mx, __shfl_xor(mx, 2, 16));
      mx = fmaxf(mx, __shfl_xor(mx, 1, 16));
      const float mnew = fmaxf(mrow[r], mx);
      const float al = __expf(mrow[r] - mnew);
      const float p0 = __expf(a - mnew);
      const float p1 = __expf(b - mnew);
      float ps = p0 + p1;
      ps += __shfl_xor(ps, 8, 16);
      ps += __shfl_xor(ps, 4, 16);
      ps += __shfl_xor(ps, 2, 16);
      ps += __shfl_xor(ps, 1, 16);
      lrow[r] = lrow[r] * al + ps;
      mrow[r] = mnew;
      p0v[r] = p0; p1v[r] = p1; alpha[r] = al;
    }
#pragma unroll
    for (int nt = 0; nt < 4; ++nt)
#pragma unroll
      for (int r = 0; r < 8; ++r) o[nt][r] *= alpha[r];

    // ---- C-layout -> A-layout transpose of P through LDS ----
#pragma unroll
    for (int r = 0; r < 8; ++r) {
      const int m = hf * 8 + r;
      pLds[m * 32 + n16]      = (bf16)p0v[r];
      pLds[m * 32 + 16 + n16] = (bf16)p1v[r];
    }
    __syncthreads();
    const v16bf pf = load_a_frag(pLds, 32);
    __syncthreads();

    // ---- O += P * V ----
#pragma unroll
    for (int nt = 0; nt < 4; ++nt) {
#if USE_ASYNC_LDS
      v16bf bv = load_b_frag(vv + (nt * 16) * 32, 32);
#else
      v16bf bv = load_b_frag(vbase + (size_t)(nt * 16) * T + key0, T);
#endif
      o[nt] = WMMA_BF16(pf, bv, o[nt]);
    }
  }

  // ---- normalize and write concat [B, T, D] ----
  const int b = bh >> 4;
  const int h = bh & (H - 1);
#pragma unroll
  for (int nt = 0; nt < 4; ++nt)
#pragma unroll
    for (int r = 0; r < 8; ++r) {
      const int m = hf * 8 + r;
      const float val = o[nt][r] / lrow[r];
      C[((size_t)b * T + qt * 16 + m) * D + h * DK + nt * 16 + n16] = (bf16)val;
    }
}

// ---------------------------------------------------------------------------
extern "C" void kernel_launch(void* const* d_in, const int* in_sizes, int n_in,
                              void* d_out, int out_size, void* d_ws, size_t ws_size,
                              hipStream_t stream) {
  const float* x  = (const float*)d_in[0];
  const float* wq = (const float*)d_in[1];
  const float* wk = (const float*)d_in[2];
  const float* wv = (const float*)d_in[3];
  const float* wo = (const float*)d_in[4];
  float* out = (float*)d_out;

  char* ws = (char*)d_ws;
  const size_t SZ_X = (size_t)M * D * sizeof(bf16);   // 8 MB
  const size_t SZ_W = (size_t)D * D * sizeof(bf16);   // 2 MB
  bf16* xb  = (bf16*)(ws);
  bf16* wqb = (bf16*)(ws + SZ_X);
  bf16* wkb = (bf16*)(ws + SZ_X + SZ_W);
  bf16* wvb = (bf16*)(ws + SZ_X + 2 * SZ_W);
  bf16* wob = (bf16*)(ws + SZ_X + 3 * SZ_W);
  bf16* qb  = (bf16*)(ws + SZ_X + 4 * SZ_W);
  bf16* kb  = (bf16*)(ws + 2 * SZ_X + 4 * SZ_W);
  bf16* vtb = (bf16*)(ws + 3 * SZ_X + 4 * SZ_W);
  bf16* cb  = xb;   // x no longer needed after projections -> reuse as concat buf

  // 1) convert inputs to bf16
  cvt_kernel<<<(M * D / 4 + 255) / 256, 256, 0, stream>>>(x, xb, M * D / 4);
  cvt_kernel<<<(D * D / 4 + 255) / 256, 256, 0, stream>>>(wq, wqb, D * D / 4);
  cvt_kernel<<<(D * D / 4 + 255) / 256, 256, 0, stream>>>(wk, wkb, D * D / 4);
  cvt_kernel<<<(D * D / 4 + 255) / 256, 256, 0, stream>>>(wv, wvb, D * D / 4);
  cvt_kernel<<<(D * D / 4 + 255) / 256, 256, 0, stream>>>(wo, wob, D * D / 4);

  // 2) q/k/v projections (v stored pre-transposed)
  const int waves  = (M / 64) * (D / 64);   // 1024 waves (64x64 tiles)
  const int blocks = waves / 8;             // 128 blocks of 8 waves
  gemm_kernel<<<blocks, 256, 0, stream>>>(xb, wqb, qb, nullptr, 0);
  gemm_kernel<<<blocks, 256, 0, stream>>>(xb, wkb, kb, nullptr, 0);
  gemm_kernel<<<blocks, 256, 0, stream>>>(xb, wvb, vtb, nullptr, 1);

  // 3) RoPE on q, k
  rope_kernel<<<(BB * H * T * 32) / 256, 256, 0, stream>>>(qb, kb);

  // 4) flash attention -> concat buffer
  attn_kernel<<<BB * H * (T / 16), 32, 0, stream>>>(qb, kb, vtb, cb);

  // 5) output projection -> fp32 d_out
  gemm_kernel<<<blocks, 256, 0, stream>>>(cb, wob, nullptr, out, 2);
}
